// BertAdapter_tensor_49933289783410
// MI455X (gfx1250) — matrix operand order repaired and checked by reading, not compiled, not run on hardware
//
#include <hip/hip_runtime.h>
#include <cstdint>
#include <cstddef>

// ---------- CDNA5 WMMA types ----------
typedef __attribute__((ext_vector_type(16))) __bf16         bf16x16;
typedef __attribute__((ext_vector_type(8)))  float          f32x8;
typedef __attribute__((ext_vector_type(16))) unsigned short u16x16;
typedef __attribute__((ext_vector_type(8)))  unsigned short u16x8;

union BF16Cast16  { u16x16 u; bf16x16 b; };
union BF16Cast2x8 { u16x8 h[2]; bf16x16 b; };

// Native RNE convert: lets the backend use v_cvt_*bf16* instead of manual bit math
__device__ __forceinline__ unsigned short f2bf(float f) {
  union { __bf16 b; unsigned short u; } c;
  c.b = (__bf16)f;
  return c.u;
}

__device__ __forceinline__ f32x8 splat8(float x) {
  f32x8 v;
#pragma unroll
  for (int j = 0; j < 8; ++j) v[j] = x;
  return v;
}

// A-fragment (16x32 bf16, MxK): this lane's 8 halves at K=cb..cb+7 and K=cb+16..cb+23
__device__ __forceinline__ bf16x16 load_a_frag(const float* __restrict__ p) {
  const float4* q0 = (const float4*)p;
  const float4* q1 = (const float4*)(p + 16);
  float4 x0 = q0[0], x1 = q0[1];
  float4 y0 = q1[0], y1 = q1[1];
  bf16x16 r;
  r[0]  = (__bf16)x0.x; r[1]  = (__bf16)x0.y; r[2]  = (__bf16)x0.z; r[3]  = (__bf16)x0.w;
  r[4]  = (__bf16)x1.x; r[5]  = (__bf16)x1.y; r[6]  = (__bf16)x1.z; r[7]  = (__bf16)x1.w;
  r[8]  = (__bf16)y0.x; r[9]  = (__bf16)y0.y; r[10] = (__bf16)y0.z; r[11] = (__bf16)y0.w;
  r[12] = (__bf16)y1.x; r[13] = (__bf16)y1.y; r[14] = (__bf16)y1.z; r[15] = (__bf16)y1.w;
  return r;
}

// B-fragment: weights are pre-swizzled so each lane reads 16 contiguous bf16 (32 B)
__device__ __forceinline__ bf16x16 load_b_frag(const unsigned short* __restrict__ p) {
  BF16Cast16 cv;
  cv.u = *(const u16x16*)p;
  return cv.b;
}

__device__ __forceinline__ f32x8 wmma_bf16(bf16x16 a, bf16x16 b, f32x8 c) {
  return __builtin_amdgcn_wmma_f32_16x16x32_bf16(false, a, false, b, (short)0, c,
                                                 false, false);
}

// ---------- TT reconstruction ----------
// cores: c0[1,8,5] c1[5,8,5] c2[5,12,5] c3[5,8,5] c4[5,8,1], rank 5
// Stage 1+2: T2[(n0*8+n1)*12+n2][r3], one block per (down/up)
__global__ __launch_bounds__(256)
void tt_stage12(const float* __restrict__ d0, const float* __restrict__ d1,
                const float* __restrict__ d2, const float* __restrict__ u0,
                const float* __restrict__ u1, const float* __restrict__ u2,
                float* __restrict__ T2d, float* __restrict__ T2u) {
  __shared__ float T1[320];  // [n0*8+n1][r2]
  const bool up = (blockIdx.x != 0);
  const float* c0 = up ? u0 : d0;
  const float* c1 = up ? u1 : d1;
  const float* c2 = up ? u2 : d2;
  float* T2 = up ? T2u : T2d;
  for (int i = threadIdx.x; i < 320; i += 256) {
    int n0 = i / 40, rem = i % 40, n1 = rem / 5, r2 = rem % 5;
    float s = 0.f;
#pragma unroll
    for (int r1 = 0; r1 < 5; ++r1) s += c0[n0 * 5 + r1] * c1[(r1 * 8 + n1) * 5 + r2];
    T1[i] = s;
  }
  __syncthreads();
  for (int i = threadIdx.x; i < 3840; i += 256) {
    int p = i / 60, rem = i % 60, n2 = rem / 5, r3 = rem % 5;
    float s = 0.f;
#pragma unroll
    for (int r2 = 0; r2 < 5; ++r2) s += T1[p * 5 + r2] * c2[(r2 * 12 + n2) * 5 + r3];
    T2[i] = s;
  }
}

// Stage 3: T3[((p*12+n2)*8+n3)][r4], 2*30720 elements
__global__ __launch_bounds__(256)
void tt_stage3(const float* __restrict__ T2d, const float* __restrict__ T2u,
               const float* __restrict__ d3, const float* __restrict__ u3,
               float* __restrict__ T3d, float* __restrict__ T3u) {
  int e = blockIdx.x * 256 + threadIdx.x;
  const bool up = (e >= 30720);
  int i = up ? e - 30720 : e;
  const float* T2 = up ? T2u : T2d;
  const float* c3 = up ? u3 : d3;
  float* T3 = up ? T3u : T3d;
  int r4 = i % 5, n3 = (i / 5) % 8, q = i / 40;
  float s = 0.f;
#pragma unroll
  for (int r3 = 0; r3 < 5; ++r3) s += T2[q * 5 + r3] * c3[(r3 * 8 + n3) * 5 + r4];
  T3[i] = s;
}

// Stage 4: final contraction + bf16 convert + WMMA-B swizzle scatter
// swizzle: block of 512 halves per (K-chunk, N-tile); within block:
//   lane = (n%16) + 16*((k%16)/8), element = (k%8) + 8*((k%32)/16)  -> 32 B contiguous per lane
__global__ __launch_bounds__(256)
void tt_stage4(const float* __restrict__ T3d, const float* __restrict__ T3u,
               const float* __restrict__ d4, const float* __restrict__ u4,
               unsigned short* __restrict__ Wd, unsigned short* __restrict__ Wu) {
  int e = blockIdx.x * 256 + threadIdx.x;
  const bool up = (e >= 49152);
  int i = up ? e - 49152 : e;
  const float* T3 = up ? T3u : T3d;
  const float* c4 = up ? u4 : d4;
  int n4 = i & 7, n3 = (i >> 3) & 7, n2 = (i >> 6) % 12;
  int n1 = (i / 768) & 7, n0 = i / 6144;
  int q = (n0 * 8 + n1) * 12 + n2;
  int t3 = (q * 8 + n3) * 5;
  float s = 0.f;
#pragma unroll
  for (int r4 = 0; r4 < 5; ++r4) s += T3[t3 + r4] * c4[r4 * 8 + n4];
  unsigned short v = f2bf(s);
  if (!up) {  // W_down[768 K][64 N], blocks ordered (kc*4 + nt)
    int k = q, n = n3 * 8 + n4;
    int kc = k >> 5, kr = k & 31, nt = n >> 4;
    int lane = (n & 15) + (((kr >> 3) & 1) << 4);
    Wd[(kc * 4 + nt) * 512 + lane * 16 + ((kr & 7) | ((kr >> 4) << 3))] = v;
  } else {    // W_up[64 K][768 N], blocks ordered (nt*2 + kc)
    int k = n0 * 8 + n1, n = n2 * 64 + n3 * 8 + n4;
    int kc = k >> 5, kr = k & 31, nt = n >> 4;
    int lane = (n & 15) + (((kr >> 3) & 1) << 4);
    Wu[(nt * 2 + kc) * 512 + lane * 16 + ((kr & 7) | ((kr >> 4) << 3))] = v;
  }
}

// ---------- fused adapter: hs -> down GEMM -> GELU -> up GEMM -> +residual ----------
// 8 waves/block, each wave owns a 32-row strip; 256 rows per block; grid = 128.
__global__ __launch_bounds__(256)
void adapter_fused(const float* __restrict__ hs, const unsigned short* __restrict__ Wd,
                   const unsigned short* __restrict__ Wu,
                   const float* __restrict__ bdown, const float* __restrict__ bup,
                   float* __restrict__ out) {
  __shared__ unsigned short actS[8][32][64];  // per-wave GELU(act) in bf16, 32 KB
  const int tid = threadIdx.x;
  const int wave = tid >> 5, lane = tid & 31;
  const int lr = lane & 15, lh = lane >> 4;
  const int rowBase = blockIdx.x * 256 + wave * 32;

  // ---- down: [32 x 768] x [768 x 64], bias preloaded into accumulators ----
  f32x8 accD[2][4];
#pragma unroll
  for (int nt = 0; nt < 4; ++nt) {
    float b = bdown[nt * 16 + lr];
    accD[0][nt] = splat8(b);
    accD[1][nt] = splat8(b);
  }
  const float* rowp0 = hs + (size_t)(rowBase + lr) * 768 + lh * 8;
  const float* rowp1 = hs + (size_t)(rowBase + 16 + lr) * 768 + lh * 8;
#pragma unroll 2
  for (int kc = 0; kc < 24; ++kc) {
    // branch-free prefetch ~4 chunks ahead (clamped to stay inside the row strip)
    int pofs = (kc < 20 ? kc + 4 : kc) * 32;
    __builtin_prefetch(rowp0 + pofs, 0, 0);
    __builtin_prefetch(rowp1 + pofs, 0, 0);
    bf16x16 a0 = load_a_frag(rowp0 + kc * 32);
    bf16x16 a1 = load_a_frag(rowp1 + kc * 32);
    const unsigned short* bp = Wd + (size_t)(kc * 4) * 512 + lane * 16;
#pragma unroll
    for (int nt = 0; nt < 4; ++nt) {
      bf16x16 bf = load_b_frag(bp + nt * 512);
      accD[0][nt] = wmma_bf16(a0, bf, accD[0][nt]);
      accD[1][nt] = wmma_bf16(a1, bf, accD[1][nt]);
    }
  }

  // ---- exact GELU, park act in LDS in row-major so we can rebuild A-fragments ----
#pragma unroll
  for (int m = 0; m < 2; ++m)
#pragma unroll
    for (int nt = 0; nt < 4; ++nt)
#pragma unroll
      for (int j = 0; j < 8; ++j) {
        float x = accD[m][nt][j];
        float g = 0.5f * x * (1.0f + erff(x * 0.70710678118654752f));
        actS[wave][m * 16 + lh * 8 + j][nt * 16 + lr] = f2bf(g);
      }
  __syncthreads();

  bf16x16 aU[2][2];
#pragma unroll
  for (int m = 0; m < 2; ++m)
#pragma unroll
    for (int k2 = 0; k2 < 2; ++k2) {
      int rrow = m * 16 + lr;
      int cb = k2 * 32 + lh * 8;
      BF16Cast2x8 cv;
      cv.h[0] = *(const u16x8*)&actS[wave][rrow][cb];
      cv.h[1] = *(const u16x8*)&actS[wave][rrow][cb + 16];
      aU[m][k2] = cv.b;
    }

  // ---- up: [32 x 64] x [64 x 768], then residual + bias, streamed out ----
  for (int nt = 0; nt < 48; ++nt) {
    float bb = bup[nt * 16 + lr];
    bf16x16 b0 = load_b_frag(Wu + (size_t)(nt * 2 + 0) * 512 + lane * 16);
    bf16x16 b1 = load_b_frag(Wu + (size_t)(nt * 2 + 1) * 512 + lane * 16);
    int col = nt * 16 + lr;
#pragma unroll
    for (int m = 0; m < 2; ++m) {
      f32x8 acc = splat8(bb);
      acc = wmma_bf16(aU[m][0], b0, acc);
      acc = wmma_bf16(aU[m][1], b1, acc);
      int rb = rowBase + m * 16 + lh * 8;
#pragma unroll
      for (int j = 0; j < 8; ++j) {
        size_t idx = (size_t)(rb + j) * 768 + col;
        // last use of hs -> NT load; out never re-read -> NT store (protect L2 set)
        float r = __builtin_nontemporal_load(hs + idx);
        __builtin_nontemporal_store(r + acc[j], out + idx);
      }
    }
  }
}

extern "C" void kernel_launch(void* const* d_in, const int* in_sizes, int n_in,
                              void* d_out, int out_size, void* d_ws, size_t ws_size,
                              hipStream_t stream) {
  (void)in_sizes; (void)n_in; (void)out_size; (void)ws_size;
  const float* hs  = (const float*)d_in[0];
  const float* bdn = (const float*)d_in[1];
  const float* bup = (const float*)d_in[2];
  const float* d0  = (const float*)d_in[3];
  const float* u0  = (const float*)d_in[4];
  const float* d1  = (const float*)d_in[5];
  const float* u1  = (const float*)d_in[6];
  const float* d2  = (const float*)d_in[7];
  const float* u2  = (const float*)d_in[8];
  const float* d3  = (const float*)d_in[9];
  const float* u3  = (const float*)d_in[10];
  const float* d4  = (const float*)d_in[11];
  const float* u4  = (const float*)d_in[12];
  float* out = (float*)d_out;

  float* ws  = (float*)d_ws;
  float* T2d = ws;           // 3840 f32
  float* T2u = ws + 3840;    // 3840 f32
  float* T3d = ws + 7680;    // 30720 f32
  float* T3u = ws + 38400;   // 30720 f32
  unsigned short* Wd = (unsigned short*)(ws + 69120);  // 49152 bf16 (swizzled)
  unsigned short* Wu = Wd + 49152;                     // 49152 bf16 (swizzled)

  tt_stage12<<<2,   256, 0, stream>>>(d0, d1, d2, u0, u1, u2, T2d, T2u);
  tt_stage3 <<<240, 256, 0, stream>>>(T2d, T2u, d3, u3, T3d, T3u);
  tt_stage4 <<<384, 256, 0, stream>>>(T3d, T3u, d4, u4, Wd, Wu);
  adapter_fused<<<128, 256, 0, stream>>>(hs, Wd, Wu, bdn, bup, out);
}
